// StateSpaceModel_40570261078520
// MI455X (gfx1250) — compile-verified, matrix operand order checked by reading, not run on hardware
//
#include <hip/hip_runtime.h>
#include <cstdint>

typedef __attribute__((ext_vector_type(16))) _Float16 v16h;
typedef __attribute__((ext_vector_type(8)))  float    v8f;

#define B_SZ     2
#define L_SZ     4096
#define D_MODEL  1024
#define D_INNER  2048
#define D_STATE  8
#define D_CONV   4
#define DT_RANK  64
#define N_LAYERS 4
#define NTOK     (B_SZ * L_SZ)   // 8192

// ---------------------------------------------------------------------------
// gfx1250 async global->LDS helpers (ASYNCcnt-tracked DMA, no VGPR staging)
// ---------------------------------------------------------------------------
__device__ __forceinline__ void async_copy_b128(const void* gaddr, uint32_t lds_off) {
    asm volatile("global_load_async_to_lds_b128 %0, %1, off"
                 :: "v"(lds_off), "v"((unsigned long long)(uintptr_t)gaddr)
                 : "memory");
}
__device__ __forceinline__ void wait_async0() {
    asm volatile("s_wait_asynccnt 0x0" ::: "memory");
}

__device__ __forceinline__ float block_reduce_sum(float v, float* red) {
    int tid = threadIdx.x;
    red[tid] = v;
    __syncthreads();
    #pragma unroll
    for (int s = 128; s > 0; s >>= 1) {
        if (tid < s) red[tid] += red[tid + s];
        __syncthreads();
    }
    float r = red[0];
    __syncthreads();
    return r;
}

// ---------------------------------------------------------------------------
// f32 -> f16 conversion (weights, once per launch)
// ---------------------------------------------------------------------------
__global__ __launch_bounds__(256) void cvt_f16_kernel(
    const float* __restrict__ in, _Float16* __restrict__ out, int n)
{
    int i = blockIdx.x * 256 + threadIdx.x;
    if (i < n) out[i] = (_Float16)in[i];
}

// ---------------------------------------------------------------------------
// Embed: feats = gelu(layernorm(source*conv_w + conv_b)); f32 + f16 outputs
// ---------------------------------------------------------------------------
__global__ __launch_bounds__(256) void embed_kernel(
    const float* __restrict__ src, const float* __restrict__ cw,
    const float* __restrict__ cb,  const float* __restrict__ g,
    const float* __restrict__ bb,  float* __restrict__ out,
    _Float16* __restrict__ outh)
{
    __shared__ float red[256];
    int tok = blockIdx.x;
    float x = src[tok];
    float vals[4];
    float s = 0.f;
    #pragma unroll
    for (int j = 0; j < 4; ++j) {
        int d = j * 256 + threadIdx.x;
        vals[j] = x * cw[d] + cb[d];
        s += vals[j];
    }
    float mu = block_reduce_sum(s, red) * (1.f / D_MODEL);
    float vs = 0.f;
    #pragma unroll
    for (int j = 0; j < 4; ++j) { float t = vals[j] - mu; vs += t * t; }
    float var = block_reduce_sum(vs, red) * (1.f / D_MODEL);
    float inv = rsqrtf(var + 1e-5f);
    #pragma unroll
    for (int j = 0; j < 4; ++j) {
        int d = j * 256 + threadIdx.x;
        float v = (vals[j] - mu) * inv * g[d] + bb[d];
        v = 0.5f * v * (1.f + erff(v * 0.70710678118654752f));
        out[(size_t)tok * D_MODEL + d]  = v;
        outh[(size_t)tok * D_MODEL + d] = (_Float16)v;
    }
}

// ---------------------------------------------------------------------------
// Tiled WMMA GEMM: C[M,N] = A[M,K] * W[N,K]^T, f16 operands, f32 accumulate.
// Block tile 128x64 (8 waves, 4Mx2N); wave tile 32x32 = 2x2 WMMA tiles.
// Double-buffered LDS staged via global_load_async_to_lds_b128.
// Requires M%128==0, N%64==0, K%32==0.  act==1: softplus(C + bias[n]).
// ---------------------------------------------------------------------------
__global__ __launch_bounds__(256) void gemm_wmma_tiled(
    const _Float16* __restrict__ A, int lda,
    const _Float16* __restrict__ W, int ldw,
    float* __restrict__ C, int ldc,
    const float* __restrict__ bias, int act,
    int N, int K)
{
    __shared__ _Float16 As[2][128][40];   // 32 halves + 8 pad (16B-aligned rows)
    __shared__ _Float16 Ws[2][64][40];

    int nbn = N >> 6;
    int bm  = blockIdx.x / nbn;
    int bn  = blockIdx.x - bm * nbn;
    int blockM = bm << 7;
    int blockN = bn << 6;

    int tid  = threadIdx.x;
    int wave = tid >> 5, lane = tid & 31;
    int wm = wave >> 1, wn = wave & 1;
    int l15 = lane & 15, hi = lane >> 4, b0 = hi << 3;

    // ---- async staging: A = 512 x 16B chunks (2/thread), W = 256 (1/thread)
    auto stage = [&](int buf, int k0) {
        #pragma unroll
        for (int i = 0; i < 2; ++i) {
            int cid = tid * 2 + i;
            int row = cid >> 2, ch = cid & 3;
            const _Float16* g = A + (size_t)(blockM + row) * lda + k0 + ch * 8;
            async_copy_b128(g, (uint32_t)(uintptr_t)&As[buf][row][ch * 8]);
        }
        {
            int row = tid >> 2, ch = tid & 3;
            const _Float16* g = W + (size_t)(blockN + row) * ldw + k0 + ch * 8;
            async_copy_b128(g, (uint32_t)(uintptr_t)&Ws[buf][row][ch * 8]);
        }
    };

    v8f acc[2][2] = {};
    int nk = K >> 5;
    stage(0, 0);
    for (int ks = 0; ks < nk; ++ks) {
        wait_async0();          // this wave's staged chunks have landed
        __syncthreads();        // everyone's chunks landed; prev buf consumed
        if (ks + 1 < nk) stage((ks + 1) & 1, (ks + 1) << 5);

        int buf = ks & 1;
        v16h a0, a1, w0, w1;
        #pragma unroll
        for (int j = 0; j < 8; ++j) {
            a0[j]     = As[buf][wm * 32 + l15][b0 + j];
            a0[j + 8] = As[buf][wm * 32 + l15][b0 + 16 + j];
            a1[j]     = As[buf][wm * 32 + 16 + l15][b0 + j];
            a1[j + 8] = As[buf][wm * 32 + 16 + l15][b0 + 16 + j];
            w0[j]     = Ws[buf][wn * 32 + l15][b0 + j];
            w0[j + 8] = Ws[buf][wn * 32 + l15][b0 + 16 + j];
            w1[j]     = Ws[buf][wn * 32 + 16 + l15][b0 + j];
            w1[j + 8] = Ws[buf][wn * 32 + 16 + l15][b0 + 16 + j];
        }
        acc[0][0] = __builtin_amdgcn_wmma_f32_16x16x32_f16(false, a0, false, w0, (short)0, acc[0][0], false, false);
        acc[0][1] = __builtin_amdgcn_wmma_f32_16x16x32_f16(false, a0, false, w1, (short)0, acc[0][1], false, false);
        acc[1][0] = __builtin_amdgcn_wmma_f32_16x16x32_f16(false, a1, false, w0, (short)0, acc[1][0], false, false);
        acc[1][1] = __builtin_amdgcn_wmma_f32_16x16x32_f16(false, a1, false, w1, (short)0, acc[1][1], false, false);
    }

    #pragma unroll
    for (int tm = 0; tm < 2; ++tm) {
        #pragma unroll
        for (int tn = 0; tn < 2; ++tn) {
            int col = blockN + wn * 32 + tn * 16 + l15;
            int rb  = blockM + wm * 32 + tm * 16 + (hi << 3);
            #pragma unroll
            for (int r = 0; r < 8; ++r) {
                float v = acc[tm][tn][r];
                if (act == 1) {
                    v += bias[col];
                    v = (v > 20.f) ? v : log1pf(__expf(v));
                }
                C[(size_t)(rb + r) * ldc + col] = v;
            }
        }
    }
}

// ---------------------------------------------------------------------------
// Simple WMMA GEMM (handles N%16 tiles, used for x_proj N=80).
// f16 operands; writes f32 C and f16 Ch.
// ---------------------------------------------------------------------------
__global__ __launch_bounds__(256) void gemm_wmma_simple(
    const _Float16* __restrict__ A, int lda,
    const _Float16* __restrict__ W, int ldw,
    float* __restrict__ C, _Float16* __restrict__ Ch, int ldc,
    int M, int N, int K)
{
    int ntilesN = N >> 4;
    int ntiles  = (M >> 4) * ntilesN;
    int wave = threadIdx.x >> 5;
    int lane = threadIdx.x & 31;
    int t = blockIdx.x * 8 + wave;
    if (t >= ntiles) return;

    int tm = t / ntilesN;
    int tn = t - tm * ntilesN;
    int l15 = lane & 15, hi = lane >> 4, b0 = hi << 3;

    const _Float16* arow = A + (size_t)((tm << 4) + l15) * lda + b0;
    const _Float16* wrow = W + (size_t)((tn << 4) + l15) * ldw + b0;

    v8f acc = {};
    for (int k0 = 0; k0 < K; k0 += 32) {
        v16h a, b;
        #pragma unroll
        for (int j = 0; j < 8; ++j) {
            a[j]     = arow[k0 + j];
            a[j + 8] = arow[k0 + 16 + j];
            b[j]     = wrow[k0 + j];
            b[j + 8] = wrow[k0 + 16 + j];
        }
        if (k0 + 32 < K) {
            __builtin_prefetch(arow + k0 + 32, 0, 3);
            __builtin_prefetch(wrow + k0 + 32, 0, 3);
        }
        acc = __builtin_amdgcn_wmma_f32_16x16x32_f16(
                  false, a, false, b, (short)0, acc, false, false);
    }

    int col = (tn << 4) + l15;
    #pragma unroll
    for (int r = 0; r < 8; ++r) {
        int row = (tm << 4) + r + (hi << 3);
        C[(size_t)row * ldc + col]  = acc[r];
        Ch[(size_t)row * ldc + col] = (_Float16)acc[r];
    }
}

// ---------------------------------------------------------------------------
// Depthwise causal conv (width 4) + bias + SiLU; f32 (scan u) + f16 (x_proj A)
// ---------------------------------------------------------------------------
__global__ __launch_bounds__(256) void conv_silu_kernel(
    const float* __restrict__ xz, const float* __restrict__ cw,
    const float* __restrict__ cb, float* __restrict__ xc,
    _Float16* __restrict__ xch)
{
    int idx = blockIdx.x * 256 + threadIdx.x;     // over NTOK * D_INNER
    int c   = idx & (D_INNER - 1);
    int tok = idx >> 11;
    int l   = tok & (L_SZ - 1);
    int b   = tok >> 12;
    float s = cb[c];
    #pragma unroll
    for (int j = 0; j < D_CONV; ++j) {
        int ls = l - (D_CONV - 1) + j;
        if (ls >= 0)
            s += cw[c * D_CONV + j] *
                 xz[(size_t)(b * L_SZ + ls) * (2 * D_INNER) + c];
    }
    float v = s / (1.f + __expf(-s));
    xc[(size_t)idx]  = v;
    xch[(size_t)idx] = (_Float16)v;
}

// ---------------------------------------------------------------------------
// Selective scan: one lane per (b,d,n); width-8 shuffle reduce over states.
// y = (scan + u*Dp) * silu(z), written as f16 (consumed only by out_proj).
// ---------------------------------------------------------------------------
__global__ __launch_bounds__(256) void scan_kernel(
    const float* __restrict__ delta, const float* __restrict__ u,
    const float* __restrict__ xdbl,  const float* __restrict__ xz,
    const float* __restrict__ A_log, const float* __restrict__ Dp,
    _Float16* __restrict__ yh)
{
    int gid = blockIdx.x * 256 + threadIdx.x;    // 32768 total
    int n = gid & (D_STATE - 1);
    int d = (gid >> 3) & (D_INNER - 1);
    int b = gid >> 14;

    float Ac = -__expf(A_log[d * D_STATE + n]);
    float dc = Dp[d];
    float h  = 0.f;
    size_t baseTok = (size_t)b * L_SZ;

    for (int l = 0; l < L_SZ; ++l) {
        size_t tok = baseTok + l;
        float dt = delta[tok * D_INNER + d];
        float uv = u[tok * D_INNER + d];
        float bn = xdbl[tok * 80 + DT_RANK + n];
        float cn = xdbl[tok * 80 + DT_RANK + D_STATE + n];
        h = h * __expf(dt * Ac) + (dt * uv) * bn;
        float yp = h * cn;
        yp += __shfl_xor(yp, 4, 8);
        yp += __shfl_xor(yp, 2, 8);
        yp += __shfl_xor(yp, 1, 8);
        if (n == 0) {
            float z  = xz[tok * (2 * D_INNER) + D_INNER + d];
            float sz = z / (1.f + __expf(-z));
            yh[tok * D_INNER + d] = (_Float16)((yp + uv * dc) * sz);
        }
    }
}

// ---------------------------------------------------------------------------
// LayerNorm over D_MODEL; f32 + f16 outputs.
// ---------------------------------------------------------------------------
__global__ __launch_bounds__(256) void layernorm_kernel(
    const float* __restrict__ x, const float* __restrict__ g,
    const float* __restrict__ bb, float* __restrict__ out,
    _Float16* __restrict__ outh)
{
    __shared__ float red[256];
    size_t base = (size_t)blockIdx.x * D_MODEL;
    float vals[4];
    float s = 0.f;
    #pragma unroll
    for (int j = 0; j < 4; ++j) {
        vals[j] = x[base + j * 256 + threadIdx.x];
        s += vals[j];
    }
    float mu = block_reduce_sum(s, red) * (1.f / D_MODEL);
    float vs = 0.f;
    #pragma unroll
    for (int j = 0; j < 4; ++j) { float t = vals[j] - mu; vs += t * t; }
    float var = block_reduce_sum(vs, red) * (1.f / D_MODEL);
    float inv = rsqrtf(var + 1e-5f);
    #pragma unroll
    for (int j = 0; j < 4; ++j) {
        int d = j * 256 + threadIdx.x;
        float v = (vals[j] - mu) * inv * g[d] + bb[d];
        out[base + d]  = v;
        outh[base + d] = (_Float16)v;
    }
}

// ---------------------------------------------------------------------------
// Decoder: sigmoid(feats . dec_w + dec_b), one block per token.
// ---------------------------------------------------------------------------
__global__ __launch_bounds__(256) void dec_kernel(
    const float* __restrict__ feats, const float* __restrict__ w,
    const float* __restrict__ bb, float* __restrict__ out)
{
    __shared__ float red[256];
    size_t base = (size_t)blockIdx.x * D_MODEL;
    float s = 0.f;
    #pragma unroll
    for (int j = 0; j < 4; ++j) {
        int d = j * 256 + threadIdx.x;
        s += feats[base + d] * w[d];
    }
    float tot = block_reduce_sum(s, red);
    if (threadIdx.x == 0)
        out[blockIdx.x] = 1.f / (1.f + __expf(-(tot + bb[0])));
}

// ---------------------------------------------------------------------------
extern "C" void kernel_launch(void* const* d_in, const int* in_sizes, int n_in,
                              void* d_out, int out_size, void* d_ws, size_t ws_size,
                              hipStream_t stream) {
    const float* source    = (const float*)d_in[0];
    const float* conv_w    = (const float*)d_in[1];
    const float* conv_b    = (const float*)d_in[2];
    const float* ln0_g     = (const float*)d_in[3];
    const float* ln0_b     = (const float*)d_in[4];
    const float* in_proj_w = (const float*)d_in[5];
    const float* conv1d_w  = (const float*)d_in[6];
    const float* conv1d_b  = (const float*)d_in[7];
    const float* x_proj_w  = (const float*)d_in[8];
    const float* dt_proj_w = (const float*)d_in[9];
    const float* dt_proj_b = (const float*)d_in[10];
    const float* A_log     = (const float*)d_in[11];
    const float* Dp        = (const float*)d_in[12];
    const float* out_proj_w= (const float*)d_in[13];
    const float* ln_g      = (const float*)d_in[14];
    const float* ln_b      = (const float*)d_in[15];
    const float* dec_w     = (const float*)d_in[16];
    const float* dec_b     = (const float*)d_in[17];
    float* out = (float*)d_out;

    // ---- workspace carve-out: f32 region, then f16 region (16B aligned) ----
    float* ws     = (float*)d_ws;
    float* feats  = ws;                                   // NTOK*1024
    float* xz     = feats  + (size_t)NTOK * D_MODEL;      // NTOK*4096
    float* xc     = xz     + (size_t)NTOK * 2 * D_INNER;  // NTOK*2048
    float* xdbl   = xc     + (size_t)NTOK * D_INNER;      // NTOK*80
    float* delta  = xdbl   + (size_t)NTOK * 80;           // NTOK*2048
    float* feats2 = delta  + (size_t)NTOK * D_INNER;      // NTOK*1024

    _Float16* hbase = (_Float16*)(feats2 + (size_t)NTOK * D_MODEL);
    _Float16* featsh = hbase;                             // NTOK*1024
    _Float16* xch    = featsh + (size_t)NTOK * D_MODEL;   // NTOK*2048
    _Float16* xdblh  = xch    + (size_t)NTOK * D_INNER;   // NTOK*80
    _Float16* ybufh  = xdblh  + (size_t)NTOK * 80;        // NTOK*2048
    _Float16* Winh   = ybufh  + (size_t)NTOK * D_INNER;   // 4*4096*1024
    _Float16* Wxph   = Winh   + (size_t)N_LAYERS * 2 * D_INNER * D_MODEL;
    _Float16* Wdth   = Wxph   + (size_t)N_LAYERS * 80 * D_INNER;
    _Float16* Wouth  = Wdth   + (size_t)N_LAYERS * D_INNER * DT_RANK;

    // ---- one-time weight conversion to f16 ----
    int n_in_w  = N_LAYERS * 2 * D_INNER * D_MODEL;
    int n_xp_w  = N_LAYERS * 80 * D_INNER;
    int n_dt_w  = N_LAYERS * D_INNER * DT_RANK;
    int n_out_w = N_LAYERS * D_MODEL * D_INNER;
    cvt_f16_kernel<<<(n_in_w  + 255) / 256, 256, 0, stream>>>(in_proj_w,  Winh,  n_in_w);
    cvt_f16_kernel<<<(n_xp_w  + 255) / 256, 256, 0, stream>>>(x_proj_w,   Wxph,  n_xp_w);
    cvt_f16_kernel<<<(n_dt_w  + 255) / 256, 256, 0, stream>>>(dt_proj_w,  Wdth,  n_dt_w);
    cvt_f16_kernel<<<(n_out_w + 255) / 256, 256, 0, stream>>>(out_proj_w, Wouth, n_out_w);

    embed_kernel<<<NTOK, 256, 0, stream>>>(source, conv_w, conv_b, ln0_g, ln0_b,
                                           feats, featsh);

    const int M = NTOK;
    for (int i = 0; i < N_LAYERS; ++i) {
        const _Float16* W_in  = Winh  + (size_t)i * 2 * D_INNER * D_MODEL;
        const _Float16* W_xp  = Wxph  + (size_t)i * 80 * D_INNER;
        const _Float16* W_dt  = Wdth  + (size_t)i * D_INNER * DT_RANK;
        const _Float16* W_out = Wouth + (size_t)i * D_MODEL * D_INNER;
        const float*    b_dt  = dt_proj_b + (size_t)i * D_INNER;

        // xz = feats @ in_proj^T   (8192 x 4096 x 1024), tiled async WMMA
        gemm_wmma_tiled<<<(M / 128) * (2 * D_INNER / 64), 256, 0, stream>>>(
            featsh, D_MODEL, W_in, D_MODEL, xz, 2 * D_INNER,
            nullptr, 0, 2 * D_INNER, D_MODEL);

        // depthwise conv + silu -> xc (f32) + xch (f16)
        conv_silu_kernel<<<(NTOK * D_INNER) / 256, 256, 0, stream>>>(
            xz, conv1d_w + (size_t)i * D_INNER * D_CONV,
            conv1d_b + (size_t)i * D_INNER, xc, xch);

        // x_dbl = xc @ x_proj^T    (8192 x 80 x 2048), N=80 -> simple kernel
        gemm_wmma_simple<<<((M >> 4) * (80 >> 4) + 7) / 8, 256, 0, stream>>>(
            xch, D_INNER, W_xp, D_INNER, xdbl, xdblh, 80, M, 80, D_INNER);

        // delta = softplus(x_dbl[:, :64] @ dt_proj^T + dt_b)  (8192 x 2048 x 64)
        gemm_wmma_tiled<<<(M / 128) * (D_INNER / 64), 256, 0, stream>>>(
            xdblh, 80, W_dt, DT_RANK, delta, D_INNER,
            b_dt, 1, D_INNER, DT_RANK);

        // selective scan (fused u*Dp + silu(z) gate) -> f16 y
        scan_kernel<<<(B_SZ * D_INNER * D_STATE) / 256, 256, 0, stream>>>(
            delta, xc, xdbl, xz,
            A_log + (size_t)i * D_INNER * D_STATE,
            Dp + (size_t)i * D_INNER, ybufh);

        // feats2 = y @ out_proj^T  (8192 x 1024 x 2048)
        gemm_wmma_tiled<<<(M / 128) * (D_MODEL / 64), 256, 0, stream>>>(
            ybufh, D_INNER, W_out, D_INNER, feats2, D_MODEL,
            nullptr, 0, D_MODEL, D_INNER);

        // per-layer layernorm -> feats (f32) + featsh (f16)
        layernorm_kernel<<<NTOK, 256, 0, stream>>>(
            feats2, ln_g + (size_t)i * D_MODEL, ln_b + (size_t)i * D_MODEL,
            feats, featsh);
    }

    dec_kernel<<<NTOK, 256, 0, stream>>>(feats, dec_w, dec_b, out);
}
